// MambaSequencePredictor_84765474554311
// MI455X (gfx1250) — matrix operand order repaired and checked by reading, not compiled, not run on hardware
//
#include <hip/hip_runtime.h>
#include <hip/hip_bf16.h>
#include <cstdint>
#include <cstddef>

// ---------------------------------------------------------------------------
// Mamba sequence predictor for MI455X (gfx1250, wave32, WMMA).
// bf16 WMMA (16x16x32, f32 accumulate) for all GEMMs with double-buffered
// async global->LDS staging (ASYNCcnt pipeline); f32 chunked parallel
// selective scan; fused conv/silu/gate elementwise kernels.
// ---------------------------------------------------------------------------

typedef __attribute__((ext_vector_type(16))) __bf16 v16bf;
typedef __attribute__((ext_vector_type(8)))  float  v8f;
typedef __attribute__((ext_vector_type(8)))  __bf16 bf16x8;
typedef int i32x4_ __attribute__((vector_size(16)));

#define B_      8
#define N_      4096
#define M_      (B_ * N_)      // 32768 rows
#define DMODEL  256
#define E_      512
#define S_      16
#define R_      16
#define KCONV   4
#define NPROJ   48             // R + 2S
#define NCHUNK  64
#define CHUNK   64
#define LLAYER  2

// ---- gfx1250 async global->LDS copy (ASYNCcnt) with safe fallback ---------
#if defined(__has_builtin)
#if __has_builtin(__builtin_amdgcn_global_load_async_to_lds_b128)
#define HAVE_ASYNC_LDS 1
#endif
#endif
#ifndef HAVE_ASYNC_LDS
#define HAVE_ASYNC_LDS 0
#endif

__device__ __forceinline__ void cp16(__bf16* lds, const __bf16* g)
{
#if HAVE_ASYNC_LDS
    // builtin signature (probed from diagnostics): (AS1 int4*, AS3 int4*, Ii, Ii)
    __bf16* gnc = (__bf16*)g;   // strip const
    __builtin_amdgcn_global_load_async_to_lds_b128(
        (__attribute__((address_space(1))) i32x4_*)gnc,
        (__attribute__((address_space(3))) i32x4_*)lds, 0, 0);
#else
    *(float4*)lds = *(const float4*)g;
#endif
}

__device__ __forceinline__ void flush_cp()
{
#if HAVE_ASYNC_LDS
#if defined(__has_builtin) && __has_builtin(__builtin_amdgcn_s_wait_asynccnt)
    __builtin_amdgcn_s_wait_asynccnt(0);
#else
    asm volatile("s_wait_asynccnt 0x0" ::: "memory");
#endif
#endif
}

__device__ __forceinline__ float sigmoidf_(float x) { return 1.0f / (1.0f + __expf(-x)); }

// ---------------------------------------------------------------------------
// Generic f32 -> bf16 cast (weights, inputs)
// ---------------------------------------------------------------------------
__global__ void cast_bf16_kernel(const float* __restrict__ in,
                                 __bf16* __restrict__ out, size_t n)
{
    size_t i = (size_t)blockIdx.x * blockDim.x + threadIdx.x;
    size_t stride = (size_t)gridDim.x * blockDim.x;
    for (; i < n; i += stride) out[i] = (__bf16)in[i];
}

// ---------------------------------------------------------------------------
// Tiled bf16 GEMM: out[M x Nmat] = A[M x K] * W^T  (W stored as [Nmat][K])
//   BIAS:  C initialized from bias[n]
//   BETA:  C initialized from existing outF (residual accumulate)
//   OUTBF: store bf16 instead of f32
// Block: 128 threads (4 waves), tile 128x64, BK=32, double-buffered LDS,
// async global->LDS staging; each wave owns 32x64 = 2x4 WMMA tiles.
// ---------------------------------------------------------------------------
template <bool BETA, bool BIAS, bool OUTBF>
__global__ __launch_bounds__(128)
void gemm_bf16_kernel(const __bf16* __restrict__ A, const __bf16* __restrict__ W,
                      const float* __restrict__ bias,
                      float* __restrict__ outF, __bf16* __restrict__ outB,
                      int Nmat, int Kdim)
{
    __shared__ __bf16 ldsA[2][128 * 40];   // rows padded to 40 halfs (80B)
    __shared__ __bf16 ldsB[2][64 * 40];    // staged as [n][k]

    const int tid  = threadIdx.x;
    const int lane = tid & 31;
    const int wave = tid >> 5;             // 0..3 : 32-row strip
    const int m0   = blockIdx.x * 128;
    const int n0   = blockIdx.y * 64;

    // staging roles: A -> one 32-elem row per thread; B -> half-row per thread
    const int brow = tid >> 1;             // 0..63
    const int bcol = (tid & 1) * 16;       // 0 / 16
    const bool bvalid = (n0 + brow) < Nmat;

    const __bf16* gA = A + (size_t)(m0 + tid) * Kdim;
    const __bf16* gW = W + (size_t)(n0 + brow) * Kdim + bcol;

    auto stage = [&](int buf, int k0) {
        __bf16* dA = &ldsA[buf][tid * 40];
        const __bf16* sA = gA + k0;
        cp16(dA + 0,  sA + 0);
        cp16(dA + 8,  sA + 8);
        cp16(dA + 16, sA + 16);
        cp16(dA + 24, sA + 24);
        __bf16* dB = &ldsB[buf][brow * 40 + bcol];
        if (bvalid) {
            const __bf16* sB = gW + k0;
            cp16(dB + 0, sB + 0);
            cp16(dB + 8, sB + 8);
        } else {
            const float4 z = make_float4(0.f, 0.f, 0.f, 0.f);
            *(float4*)(dB + 0) = z;
            *(float4*)(dB + 8) = z;
        }
#if !HAVE_ASYNC_LDS
        if (k0 + 32 < Kdim) __builtin_prefetch(sA + 32, 0, 1);
#endif
    };

    // ---- accumulator init ----
    v8f acc[2][4];
#pragma unroll
    for (int i = 0; i < 2; ++i)
#pragma unroll
        for (int j = 0; j < 4; ++j) {
            const int n = n0 + j * 16 + (lane & 15);
            if (BIAS) {
                const float bv = (n < Nmat) ? bias[n] : 0.0f;
#pragma unroll
                for (int r = 0; r < 8; ++r) acc[i][j][r] = bv;
            } else if (BETA) {
                const int mb = m0 + wave * 32 + i * 16 + ((lane >> 4) << 3);
#pragma unroll
                for (int r = 0; r < 8; ++r)
                    acc[i][j][r] = (n < Nmat) ? outF[(size_t)(mb + r) * Nmat + n] : 0.0f;
            } else {
#pragma unroll
                for (int r = 0; r < 8; ++r) acc[i][j][r] = 0.0f;
            }
        }

    // ---- software-pipelined K loop ----
    stage(0, 0);
    flush_cp();
    __syncthreads();

    const int niter = Kdim >> 5;           // K is a multiple of 32
    for (int it = 0; it < niter; ++it) {
        const int cur = it & 1;
        if (it + 1 < niter) stage(cur ^ 1, (it + 1) << 5);  // overlap with compute

        // fragments (documented gfx1250 bf16 WMMA lane layout)
        v16bf afrag[2], bfrag[4];
#pragma unroll
        for (int i = 0; i < 2; ++i) {
            const int r = wave * 32 + i * 16 + (lane & 15);
            const int g = (lane >> 4) << 3;   // lanes 0-15: K 0-7/16-23; 16-31: 8-15/24-31
            const __bf16* p = &ldsA[cur][r * 40];
            v16bf f;
            ((float4*)&f)[0] = *(const float4*)(p + g);
            ((float4*)&f)[1] = *(const float4*)(p + 16 + g);
            afrag[i] = f;
        }
#pragma unroll
        for (int j = 0; j < 4; ++j) {
            const int c  = j * 16 + (lane & 15);
            const int g2 = (lane >> 4) << 3;  // lanes 0-15: K 0-7/16-23; 16-31: 8-15/24-31
            const __bf16* p = &ldsB[cur][c * 40];
            v16bf f;
            ((float4*)&f)[0] = *(const float4*)(p + g2);
            ((float4*)&f)[1] = *(const float4*)(p + 16 + g2);
            bfrag[j] = f;
        }

#pragma unroll
        for (int i = 0; i < 2; ++i)
#pragma unroll
            for (int j = 0; j < 4; ++j)
                acc[i][j] = __builtin_amdgcn_wmma_f32_16x16x32_bf16(
                    false, afrag[i], false, bfrag[j], (short)0, acc[i][j], false, false);

        flush_cp();       // next tile resident in LDS
        __syncthreads();  // all waves done reading `cur`, `nxt` published
    }

    // ---- store ----
#pragma unroll
    for (int i = 0; i < 2; ++i)
#pragma unroll
        for (int j = 0; j < 4; ++j) {
            const int n = n0 + j * 16 + (lane & 15);
            if (n >= Nmat) continue;
            const int mb = m0 + wave * 32 + i * 16 + ((lane >> 4) << 3);
#pragma unroll
            for (int r = 0; r < 8; ++r) {
                const float v = acc[i][j][r];
                if (OUTBF) outB[(size_t)(mb + r) * Nmat + n] = (__bf16)v;
                else       outF[(size_t)(mb + r) * Nmat + n] = v;
            }
        }
}

// ---------------------------------------------------------------------------
// LayerNorm over D_MODEL=256, one wave per row, bf16 output.
// ---------------------------------------------------------------------------
__global__ __launch_bounds__(256)
void ln_kernel(const float* __restrict__ h, const float* __restrict__ g,
               const float* __restrict__ b, __bf16* __restrict__ u)
{
    const int lane = threadIdx.x & 31;
    const int wave = threadIdx.x >> 5;
    const int row  = blockIdx.x * 8 + wave;

    const float* hp = h + (size_t)row * DMODEL + lane * 8;
    const float4 v0 = *(const float4*)(hp);
    const float4 v1 = *(const float4*)(hp + 4);
    float va[8] = {v0.x, v0.y, v0.z, v0.w, v1.x, v1.y, v1.z, v1.w};

    float s = 0.f, s2 = 0.f;
#pragma unroll
    for (int r = 0; r < 8; ++r) { s += va[r]; s2 += va[r] * va[r]; }
#pragma unroll
    for (int m = 16; m; m >>= 1) {
        s  += __shfl_xor(s, m);
        s2 += __shfl_xor(s2, m);
    }
    const float mu  = s * (1.0f / DMODEL);
    const float var = s2 * (1.0f / DMODEL) - mu * mu;
    const float inv = rsqrtf(var + 1e-5f);

    const float* gp = g + lane * 8;
    const float* bp = b + lane * 8;
    bf16x8 o;
#pragma unroll
    for (int r = 0; r < 8; ++r)
        o[r] = (__bf16)((va[r] - mu) * inv * gp[r] + bp[r]);
    *(bf16x8*)(u + (size_t)row * DMODEL + lane * 8) = o;
}

// ---------------------------------------------------------------------------
// Causal depthwise conv (K=4) + bias + SiLU over xc half of xz.
// ---------------------------------------------------------------------------
__global__ __launch_bounds__(512)
void conv_kernel(const __bf16* __restrict__ xz, const float* __restrict__ cw,
                 const float* __restrict__ cb, __bf16* __restrict__ xcs,
                 float* __restrict__ xs)
{
    const int e   = threadIdx.x;
    const int rid = blockIdx.x;
    const int t   = rid & (N_ - 1);

    const float w0 = cw[e * 4 + 0], w1 = cw[e * 4 + 1];
    const float w2 = cw[e * 4 + 2], w3 = cw[e * 4 + 3];

    const size_t stride = 2 * E_;
    const __bf16* col = xz + e;
    const size_t r0 = (size_t)rid;

    float acc = cb[e];
    if (t >= 3) acc += w0 * (float)col[(r0 - 3) * stride];
    if (t >= 2) acc += w1 * (float)col[(r0 - 2) * stride];
    if (t >= 1) acc += w2 * (float)col[(r0 - 1) * stride];
    acc += w3 * (float)col[r0 * stride];

    const float sv = acc * sigmoidf_(acc);
    xcs[(size_t)rid * E_ + e] = (__bf16)sv;
    xs [(size_t)rid * E_ + e] = sv;
}

// ---------------------------------------------------------------------------
// dt = softplus(proj[:, :R] @ dt_w.T + dt_b). One block per row, thread = e.
// ---------------------------------------------------------------------------
__global__ __launch_bounds__(512)
void dt_kernel(const float* __restrict__ proj, const float* __restrict__ dtw,
               const float* __restrict__ dtb, float* __restrict__ dt)
{
    __shared__ float p16[R_];
    const int e   = threadIdx.x;
    const int rid = blockIdx.x;
    if (e < R_) p16[e] = proj[(size_t)rid * NPROJ + e];
    __syncthreads();

    float acc = dtb[e];
#pragma unroll
    for (int r = 0; r < R_; ++r) acc += p16[r] * dtw[e * R_ + r];
    const float sp = (acc > 20.0f) ? acc : log1pf(__expf(acc));
    dt[(size_t)rid * E_ + e] = sp;
}

// ---------------------------------------------------------------------------
// Chunked selective scan (diagonal SSM recurrence).
// Wave task = (b, chunk, e-pair); lane = (e&1)*16 + s.
// ---------------------------------------------------------------------------
__global__ __launch_bounds__(256)
void scan_p1(const float* __restrict__ dt, const float* __restrict__ xs,
             const float* __restrict__ proj, const float* __restrict__ alog,
             float* __restrict__ carry, float* __restrict__ sumdt)
{
    const int lane  = threadIdx.x & 31;
    const int wid   = blockIdx.x * 8 + (threadIdx.x >> 5);
    const int pair  = wid & 255;
    const int chunk = (wid >> 8) & 63;
    const int b     = wid >> 14;
    const int e     = pair * 2 + (lane >> 4);
    const int s     = lane & 15;

    const float Ael = -__expf(alog[e * S_ + s]);
    float h = 0.0f, sd = 0.0f;

    size_t row = (size_t)b * N_ + (size_t)chunk * CHUNK;
    for (int tt = 0; tt < CHUNK; ++tt, ++row) {
        const float dte = dt[row * E_ + e];
        const float xe  = xs[row * E_ + e];
        const float Bts = proj[row * NPROJ + R_ + s];
        h = h * __expf(dte * Ael) + dte * xe * Bts;
        sd += dte;
    }
    const size_t ci = (((size_t)b * NCHUNK + chunk) * E_ + e) * S_ + s;
    carry[ci] = h;
    if (s == 0) sumdt[((size_t)b * NCHUNK + chunk) * E_ + e] = sd;
}

__global__ __launch_bounds__(256)
void scan_p2(const float* __restrict__ alog, float* __restrict__ carry,
             const float* __restrict__ sumdt)
{
    const int id = blockIdx.x * 256 + threadIdx.x;   // 0 .. B*E*S-1
    const int s  = id & 15;
    const int e  = (id >> 4) & (E_ - 1);
    const int b  = id >> 13;

    const float Ael = -__expf(alog[e * S_ + s]);
    float H = 0.0f;
    for (int c = 0; c < NCHUNK; ++c) {
        const size_t ci = (((size_t)b * NCHUNK + c) * E_ + e) * S_ + s;
        const float local = carry[ci];
        carry[ci] = H;                                          // carry-in for chunk c
        H = __expf(Ael * sumdt[((size_t)b * NCHUNK + c) * E_ + e]) * H + local;
    }
}

__global__ __launch_bounds__(256)
void scan_p3(const float* __restrict__ dt, const float* __restrict__ xs,
             const float* __restrict__ proj, const float* __restrict__ alog,
             const float* __restrict__ carry, const float* __restrict__ Dp,
             const __bf16* __restrict__ xz, __bf16* __restrict__ gout)
{
    const int lane  = threadIdx.x & 31;
    const int wid   = blockIdx.x * 8 + (threadIdx.x >> 5);
    const int pair  = wid & 255;
    const int chunk = (wid >> 8) & 63;
    const int b     = wid >> 14;
    const int e     = pair * 2 + (lane >> 4);
    const int s     = lane & 15;

    const float Ael = -__expf(alog[e * S_ + s]);
    const float De  = Dp[e];
    const size_t ci = (((size_t)b * NCHUNK + chunk) * E_ + e) * S_ + s;
    float h = carry[ci];

    size_t row = (size_t)b * N_ + (size_t)chunk * CHUNK;
    for (int tt = 0; tt < CHUNK; ++tt, ++row) {
        const float dte = dt[row * E_ + e];
        const float xe  = xs[row * E_ + e];
        const float Bts = proj[row * NPROJ + R_ + s];
        const float Cts = proj[row * NPROJ + R_ + S_ + s];
        h = h * __expf(dte * Ael) + dte * xe * Bts;

        float v = h * Cts;                       // reduce over s within 16 lanes
        v += __shfl_xor(v, 1, 16);
        v += __shfl_xor(v, 2, 16);
        v += __shfl_xor(v, 4, 16);
        v += __shfl_xor(v, 8, 16);

        if (s == 0) {
            const float y  = v + De * xe;
            const float zv = (float)xz[row * (2 * E_) + E_ + e];
            const float gg = y * zv * sigmoidf_(zv);
            gout[row * E_ + e] = (__bf16)gg;
        }
    }
}

// ---------------------------------------------------------------------------
// Launch
// ---------------------------------------------------------------------------
extern "C" void kernel_launch(void* const* d_in, const int* in_sizes, int n_in,
                              void* d_out, int out_size, void* d_ws, size_t ws_size,
                              hipStream_t stream)
{
    const float* x       = (const float*)d_in[0];
    const float* W_in    = (const float*)d_in[1];
    const float* b_in    = (const float*)d_in[2];
    const float* ln_g    = (const float*)d_in[3];
    const float* ln_b    = (const float*)d_in[4];
    const float* in_w    = (const float*)d_in[5];
    const float* conv_w  = (const float*)d_in[6];
    const float* conv_b  = (const float*)d_in[7];
    const float* xproj_w = (const float*)d_in[8];
    const float* dt_w    = (const float*)d_in[9];
    const float* dt_b    = (const float*)d_in[10];
    const float* A_log   = (const float*)d_in[11];
    const float* Dp      = (const float*)d_in[12];
    const float* out_w   = (const float*)d_in[13];
    const float* W_out   = (const float*)d_in[14];
    const float* b_out   = (const float*)d_in[15];
    float* out = (float*)d_out;
    (void)in_sizes; (void)n_in; (void)out_size; (void)ws_size;

    char* ws = (char*)d_ws;
    size_t off = 0;
    auto alloc = [&](size_t bytes) -> char* {
        char* p = ws + off;
        off += (bytes + 255) & ~(size_t)255;
        return p;
    };

    float*  hbuf  = (float*) alloc((size_t)M_ * DMODEL * 4);          // persistent h
    __bf16* shbuf = (__bf16*)alloc((size_t)M_ * E_ * 2);              // shared: u / gated / h_bf16
    __bf16* xzb   = (__bf16*)alloc((size_t)M_ * 2 * E_ * 2);          // xz (bf16)
    __bf16* xcsb  = (__bf16*)alloc((size_t)M_ * E_ * 2);              // conv+silu (bf16)
    float*  xsf   = (float*) alloc((size_t)M_ * E_ * 4);              // conv+silu (f32)
    float*  projf = (float*) alloc((size_t)M_ * NPROJ * 4);
    float*  dtf   = (float*) alloc((size_t)M_ * E_ * 4);
    float*  carry = (float*) alloc((size_t)B_ * NCHUNK * E_ * S_ * 4);
    float*  sumdt = (float*) alloc((size_t)B_ * NCHUNK * E_ * 4);
    __bf16* xb    = (__bf16*)alloc((size_t)M_ * 64 * 2);
    __bf16* wWin  = (__bf16*)alloc((size_t)DMODEL * 64 * 2);
    __bf16* wInw  = (__bf16*)alloc((size_t)LLAYER * 2 * E_ * DMODEL * 2);
    __bf16* wXp   = (__bf16*)alloc((size_t)LLAYER * NPROJ * E_ * 2);
    __bf16* wOw   = (__bf16*)alloc((size_t)LLAYER * DMODEL * E_ * 2);
    __bf16* wWout = (__bf16*)alloc((size_t)32 * DMODEL * 2);

    auto cast = [&](const float* src, __bf16* dst, size_t n) {
        int blocks = (int)((n + 1023) / 1024);
        if (blocks > 4096) blocks = 4096;
        if (blocks < 1)    blocks = 1;
        cast_bf16_kernel<<<blocks, 256, 0, stream>>>(src, dst, n);
    };
    cast(x,       xb,    (size_t)M_ * 64);
    cast(W_in,    wWin,  (size_t)DMODEL * 64);
    cast(in_w,    wInw,  (size_t)LLAYER * 2 * E_ * DMODEL);
    cast(xproj_w, wXp,   (size_t)LLAYER * NPROJ * E_);
    cast(out_w,   wOw,   (size_t)LLAYER * DMODEL * E_);
    cast(W_out,   wWout, (size_t)32 * DMODEL);

    const dim3 blk(128);
    const int scan_blocks = (B_ * NCHUNK * (E_ / 2)) / 8;   // 8 waves/block

    // h = x @ W_in.T + b_in
    gemm_bf16_kernel<false, true, false><<<dim3(M_ / 128, DMODEL / 64), blk, 0, stream>>>(
        xb, wWin, b_in, hbuf, nullptr, DMODEL, 64);

    for (int l = 0; l < LLAYER; ++l) {
        // u = LayerNorm(h)
        ln_kernel<<<M_ / 8, 256, 0, stream>>>(hbuf, ln_g + l * DMODEL, ln_b + l * DMODEL, shbuf);
        // xz = u @ in_w.T  (bf16 out)
        gemm_bf16_kernel<false, false, true><<<dim3(M_ / 128, (2 * E_) / 64), blk, 0, stream>>>(
            shbuf, wInw + (size_t)l * 2 * E_ * DMODEL, nullptr, nullptr, xzb, 2 * E_, DMODEL);
        // xc = silu(conv(xc) + b)
        conv_kernel<<<M_, E_, 0, stream>>>(
            xzb, conv_w + (size_t)l * E_ * KCONV, conv_b + (size_t)l * E_, xcsb, xsf);
        // proj = xc @ xproj_w.T (48 cols)
        gemm_bf16_kernel<false, false, false><<<dim3(M_ / 128, 1), blk, 0, stream>>>(
            xcsb, wXp + (size_t)l * NPROJ * E_, nullptr, projf, nullptr, NPROJ, E_);
        // dt = softplus(proj[:, :R] @ dt_w.T + dt_b)
        dt_kernel<<<M_, E_, 0, stream>>>(
            projf, dt_w + (size_t)l * E_ * R_, dt_b + (size_t)l * E_, dtf);
        // chunked selective scan + D-skip + silu(z) gate -> shbuf (bf16)
        const float* alog = A_log + (size_t)l * E_ * S_;
        scan_p1<<<scan_blocks, 256, 0, stream>>>(dtf, xsf, projf, alog, carry, sumdt);
        scan_p2<<<(B_ * E_ * S_) / 256, 256, 0, stream>>>(alog, carry, sumdt);
        scan_p3<<<scan_blocks, 256, 0, stream>>>(
            dtf, xsf, projf, alog, carry, Dp + (size_t)l * E_, xzb, shbuf);
        // h += gated @ out_w.T   (beta = 1)
        gemm_bf16_kernel<true, false, false><<<dim3(M_ / 128, DMODEL / 64), blk, 0, stream>>>(
            shbuf, wOw + (size_t)l * DMODEL * E_, nullptr, hbuf, nullptr, DMODEL, E_);
    }

    // out = h @ W_out.T + b_out
    cast(hbuf, shbuf, (size_t)M_ * DMODEL);
    gemm_bf16_kernel<false, true, false><<<dim3(M_ / 128, 1), blk, 0, stream>>>(
        shbuf, wWout, b_out, out, nullptr, 32, DMODEL);
}
